// S6Block_16320875724913
// MI455X (gfx1250) — compile-verified, matrix operand order checked by reading, not run on hardware
//
#include <hip/hip_runtime.h>
#include <hip/hip_bf16.h>

// ---------------------------------------------------------------------------
// Mamba / S6 block for gfx1250 (MI455X), wave32 + WMMA bf16 (f32 accumulate).
// B=4, L=1024, D_MODEL=512, D_INNER=1024, D_STATE=16, DT_RANK=128.
// GEMMs use register-blocked WMMA (WMxWN tiles of 16x16 per wave) to raise
// arithmetic intensity to ~32 FLOP/B so the matrix pipes, not L2, bound.
// ---------------------------------------------------------------------------

typedef __attribute__((ext_vector_type(16))) __bf16 v16bf;
typedef __attribute__((ext_vector_type(8)))  __bf16 v8bf;
typedef __attribute__((ext_vector_type(8)))  float  v8f;

#define TOKENS   4096
#define DMODEL   512
#define DINNER   1024
#define DSTATE   16
#define DTRANK   128
#define LSEQ     1024
#define BSZ      4

// -------------------- generic f32 -> bf16 convert --------------------------
__global__ void cvt_bf16_kernel(const float* __restrict__ in,
                                __bf16* __restrict__ out, int n) {
  int i = blockIdx.x * 256 + threadIdx.x;
  if (i < n) out[i] = (__bf16)in[i];
}

// -------------------- LayerNorm (one wave per token) -----------------------
__global__ __launch_bounds__(256)
void ln_kernel(const float* __restrict__ x, const float* __restrict__ g,
               const float* __restrict__ b, __bf16* __restrict__ xn) {
  const int token = blockIdx.x * 8 + (threadIdx.x >> 5);
  const int lane  = threadIdx.x & 31;
  const float* row = x + (size_t)token * DMODEL;
  float vals[16];
  float s = 0.f, sq = 0.f;
#pragma unroll
  for (int i = 0; i < 16; ++i) {
    float v = row[lane + i * 32];
    vals[i] = v; s += v; sq += v * v;
  }
#pragma unroll
  for (int off = 16; off > 0; off >>= 1) {
    s  += __shfl_xor(s,  off, 32);
    sq += __shfl_xor(sq, off, 32);
  }
  const float mu   = s  * (1.f / DMODEL);
  const float var  = sq * (1.f / DMODEL) - mu * mu;
  const float rstd = rsqrtf(var + 1e-5f);
#pragma unroll
  for (int i = 0; i < 16; ++i) {
    const int c = lane + i * 32;
    const float v = (vals[i] - mu) * rstd * g[c] + b[c];
    xn[(size_t)token * DMODEL + c] = (__bf16)v;
  }
}

// -------------------- WMMA GEMM:  C[M,N] = A[M,K] * W[N,K]^T ---------------
// Register-blocked: each wave owns a (WM*16) x (WN*16) macro-tile; per K=32
// step it loads WM A-fragments + WN B-fragments (two b128 loads each, per the
// CDNA5 16-bit fragment layout) and issues WM*WN back-to-back WMMAs.
// Tile predicate is wave-uniform -> EXEC all-ones at every WMMA.
template <int WM, int WN>
__global__ __launch_bounds__(256)
void wmma_gemm_bf16(const __bf16* __restrict__ A, const __bf16* __restrict__ W,
                    float* __restrict__ C, const float* __restrict__ addend,
                    int M, int N, int K) {
  const int lane   = threadIdx.x & 31;
  const int wave   = threadIdx.x >> 5;
  const int gn     = (N >> 4) / WN;                  // macro-tiles along N
  const int wid    = blockIdx.x * 8 + wave;          // wave-uniform
  const int total  = ((M >> 4) / WM) * gn;
  if (wid >= total) return;                          // uniform exit
  const int wm = (wid / gn) * (WM * 16);
  const int wn = (wid % gn) * (WN * 16);
  const int hi = lane >> 4;                          // half-wave select
  const int lo = lane & 15;
  const int koff = hi * 8;

  v8f zero = {};
  v8f acc[WM][WN];
#pragma unroll
  for (int i = 0; i < WM; ++i)
#pragma unroll
    for (int j = 0; j < WN; ++j) acc[i][j] = zero;

  union Frag { v16bf v; v8bf h[2]; };

  for (int k0 = 0; k0 < K; k0 += 32) {
    Frag af[WM], bf[WN];
#pragma unroll
    for (int i = 0; i < WM; ++i) {
      const __bf16* p = A + (size_t)(wm + i * 16 + lo) * K + k0 + koff;
      af[i].h[0] = *(const v8bf*)p;
      af[i].h[1] = *(const v8bf*)(p + 16);
      if (k0 + 32 < K) __builtin_prefetch(p + 32, 0, 3);
    }
#pragma unroll
    for (int j = 0; j < WN; ++j) {
      const __bf16* p = W + (size_t)(wn + j * 16 + lo) * K + k0 + koff;
      bf[j].h[0] = *(const v8bf*)p;
      bf[j].h[1] = *(const v8bf*)(p + 16);
      if (k0 + 32 < K) __builtin_prefetch(p + 32, 0, 3);
    }
#pragma unroll
    for (int i = 0; i < WM; ++i)
#pragma unroll
      for (int j = 0; j < WN; ++j)
        acc[i][j] = __builtin_amdgcn_wmma_f32_16x16x32_bf16(
            false, af[i].v, false, bf[j].v, (short)0, acc[i][j], false, false);
  }

#pragma unroll
  for (int i = 0; i < WM; ++i)
#pragma unroll
    for (int j = 0; j < WN; ++j) {
      const int cn = wn + j * 16 + lo;
#pragma unroll
      for (int v = 0; v < 8; ++v) {
        const int cm = wm + i * 16 + v + hi * 8;     // C/D layout (ISA 7.12.2)
        const size_t idx = (size_t)cm * N + cn;
        float r = acc[i][j][v];
        if (addend) r += addend[idx];
        C[idx] = r;
      }
    }
}

// -------------------- SiLU split: xi = silu(xz[:, :1024]) ------------------
__global__ void silu_split_kernel(const float* __restrict__ xz,
                                  float* __restrict__ xi,
                                  __bf16* __restrict__ xib, int total) {
  int i = blockIdx.x * 256 + threadIdx.x;
  if (i >= total) return;
  const int r = i >> 10, c = i & (DINNER - 1);
  const float v  = xz[(size_t)r * (2 * DINNER) + c];
  const float sv = v / (1.f + __expf(-v));
  xi[i]  = sv;
  xib[i] = (__bf16)sv;
}

// -------------------- slice dt_raw = ssm[:, :128] -> bf16 ------------------
__global__ void slice_cvt_kernel(const float* __restrict__ ssm,
                                 __bf16* __restrict__ out, int total) {
  int i = blockIdx.x * 256 + threadIdx.x;
  if (i >= total) return;
  const int r = i >> 7, c = i & (DTRANK - 1);
  out[i] = (__bf16)ssm[(size_t)r * (DTRANK + 2 * DSTATE) + c];
}

// -------------------- dt = softplus(dt_pre + b_dt)  (in place) -------------
__global__ void dt_softplus_kernel(float* __restrict__ dt,
                                   const float* __restrict__ bdt, int total) {
  int i = blockIdx.x * 256 + threadIdx.x;
  if (i >= total) return;
  const float v = dt[i] + bdt[i & (DINNER - 1)];
  dt[i] = (v > 20.f) ? v : log1pf(__expf(v));
}

// -------------------- selective scan (sequential over t) -------------------
// 4096 channels (b,d); one thread each; h[16] in registers; per-step B/C
// vectors staged in LDS (shared by all 256 channels of the block).
// Fuses + D*x, * silu(z) and the bf16 cast for the output GEMM.
__global__ __launch_bounds__(256)
void scan_kernel(const float* __restrict__ ssm, const float* __restrict__ dt,
                 const float* __restrict__ xi, const float* __restrict__ xz,
                 const float* __restrict__ A_log, const float* __restrict__ Dp,
                 __bf16* __restrict__ yb) {
  const int b = blockIdx.x >> 2;
  const int d = ((blockIdx.x & 3) << 8) + threadIdx.x;
  float A[DSTATE], h[DSTATE];
#pragma unroll
  for (int s = 0; s < DSTATE; ++s) {
    A[s] = -__expf(A_log[(size_t)d * DSTATE + s]);
    h[s] = 0.f;
  }
  const float Dd = Dp[d];
  __shared__ float sB[DSTATE], sC[DSTATE];
  for (int t = 0; t < LSEQ; ++t) {
    const size_t row = (size_t)b * LSEQ + t;
    if (threadIdx.x < 32) {
      const float v = ssm[row * (DTRANK + 2 * DSTATE) + DTRANK + threadIdx.x];
      if (threadIdx.x < DSTATE) sB[threadIdx.x] = v;
      else                      sC[threadIdx.x - DSTATE] = v;
    }
    __syncthreads();
    const float dtv = dt[row * DINNER + d];
    const float xv  = xi[row * DINNER + d];
    const float zv  = xz[row * (2 * DINNER) + DINNER + d];
    const float dBx = dtv * xv;
    float acc = 0.f;
#pragma unroll
    for (int s = 0; s < DSTATE; ++s) {
      h[s] = __expf(dtv * A[s]) * h[s] + dBx * sB[s];
      acc += h[s] * sC[s];
    }
    const float yv = (acc + Dd * xv) * (zv / (1.f + __expf(-zv)));
    yb[row * DINNER + d] = (__bf16)yv;
    __syncthreads();
  }
}

// ---------------------------------------------------------------------------
extern "C" void kernel_launch(void* const* d_in, const int* in_sizes, int n_in,
                              void* d_out, int out_size, void* d_ws, size_t ws_size,
                              hipStream_t stream) {
  const float* x     = (const float*)d_in[0];
  const float* W_in  = (const float*)d_in[1];
  const float* W_xp  = (const float*)d_in[2];
  const float* W_dt  = (const float*)d_in[3];
  const float* b_dt  = (const float*)d_in[4];
  const float* A_log = (const float*)d_in[5];
  const float* Dvec  = (const float*)d_in[6];
  const float* W_out = (const float*)d_in[7];
  const float* ln_g  = (const float*)d_in[8];
  const float* ln_b  = (const float*)d_in[9];
  float* out = (float*)d_out;

  // ---- workspace carve-up (256B aligned) ----
  char* base = (char*)d_ws;
  size_t off = 0;
  auto alloc = [&](size_t bytes) -> char* {
    char* p = base + off;
    off += (bytes + 255) & ~(size_t)255;
    return p;
  };
  __bf16* xn_bf   = (__bf16*)alloc((size_t)TOKENS * DMODEL * 2);
  __bf16* Win_bf  = (__bf16*)alloc((size_t)2 * DINNER * DMODEL * 2);
  __bf16* Wxp_bf  = (__bf16*)alloc((size_t)(DTRANK + 2 * DSTATE) * DINNER * 2);
  __bf16* Wdt_bf  = (__bf16*)alloc((size_t)DINNER * DTRANK * 2);
  __bf16* Wout_bf = (__bf16*)alloc((size_t)DMODEL * DINNER * 2);
  float*  xz      = (float*)alloc((size_t)TOKENS * 2 * DINNER * 4);
  float*  xi      = (float*)alloc((size_t)TOKENS * DINNER * 4);
  __bf16* xi_bf   = (__bf16*)alloc((size_t)TOKENS * DINNER * 2);
  float*  ssm     = (float*)alloc((size_t)TOKENS * (DTRANK + 2 * DSTATE) * 4);
  __bf16* dtr_bf  = (__bf16*)alloc((size_t)TOKENS * DTRANK * 2);
  float*  dt      = (float*)alloc((size_t)TOKENS * DINNER * 4);
  __bf16* y_bf    = (__bf16*)alloc((size_t)TOKENS * DINNER * 2);
  (void)ws_size; (void)n_in; (void)in_sizes; (void)out_size;

  // ---- 0: weights -> bf16 ----
  {
    int n1 = 2 * DINNER * DMODEL;
    cvt_bf16_kernel<<<(n1 + 255) / 256, 256, 0, stream>>>(W_in, Win_bf, n1);
    int n2 = (DTRANK + 2 * DSTATE) * DINNER;
    cvt_bf16_kernel<<<(n2 + 255) / 256, 256, 0, stream>>>(W_xp, Wxp_bf, n2);
    int n3 = DINNER * DTRANK;
    cvt_bf16_kernel<<<(n3 + 255) / 256, 256, 0, stream>>>(W_dt, Wdt_bf, n3);
    int n4 = DMODEL * DINNER;
    cvt_bf16_kernel<<<(n4 + 255) / 256, 256, 0, stream>>>(W_out, Wout_bf, n4);
  }

  // ---- 1: LayerNorm -> bf16 activations ----
  ln_kernel<<<TOKENS / 8, 256, 0, stream>>>(x, ln_g, ln_b, xn_bf);

  // ---- 2: xz = xn @ W_in^T   [4096 x 2048 x 512], 4x4 blocking ----
  {
    int waves = ((TOKENS / 16) / 4) * ((2 * DINNER / 16) / 4);   // 2048
    wmma_gemm_bf16<4, 4><<<waves / 8, 256, 0, stream>>>(
        xn_bf, Win_bf, xz, nullptr, TOKENS, 2 * DINNER, DMODEL);
  }

  // ---- 3: x_inner = silu(xz[:, :1024]) ----
  {
    int n = TOKENS * DINNER;
    silu_split_kernel<<<n / 256, 256, 0, stream>>>(xz, xi, xi_bf, n);
  }

  // ---- 4: ssm = x_inner @ W_xp^T   [4096 x 160 x 1024], 4x1 blocking ----
  {
    int waves = ((TOKENS / 16) / 4) * ((DTRANK + 2 * DSTATE) / 16); // 640
    wmma_gemm_bf16<4, 1><<<waves / 8, 256, 0, stream>>>(
        xi_bf, Wxp_bf, ssm, nullptr, TOKENS, DTRANK + 2 * DSTATE, DINNER);
  }

  // ---- 5: dt_raw slice -> bf16; dt_pre = dt_raw @ W_dt^T [4096x1024x128] ----
  {
    int n = TOKENS * DTRANK;
    slice_cvt_kernel<<<n / 256, 256, 0, stream>>>(ssm, dtr_bf, n);
    int waves = ((TOKENS / 16) / 4) * ((DINNER / 16) / 4);       // 1024
    wmma_gemm_bf16<4, 4><<<waves / 8, 256, 0, stream>>>(
        dtr_bf, Wdt_bf, dt, nullptr, TOKENS, DINNER, DTRANK);
    int m = TOKENS * DINNER;
    dt_softplus_kernel<<<m / 256, 256, 0, stream>>>(dt, b_dt, m);
  }

  // ---- 6: selective scan (fuses +D*x, *silu(z), bf16 cast) ----
  scan_kernel<<<BSZ * (DINNER / 256), 256, 0, stream>>>(ssm, dt, xi, xz,
                                                        A_log, Dvec, y_bf);

  // ---- 7: out = y @ W_out^T + residual  [4096 x 512 x 1024], 4x4 blocking ----
  {
    int waves = ((TOKENS / 16) / 4) * ((DMODEL / 16) / 4);       // 512
    wmma_gemm_bf16<4, 4><<<waves / 8, 256, 0, stream>>>(
        y_bf, Wout_bf, out, x, TOKENS, DMODEL, DINNER);
  }
}
